// MultiHead_29394756174033
// MI455X (gfx1250) — matrix operand order repaired and checked by reading, not compiled, not run on hardware
//
#include <hip/hip_runtime.h>
#include <hip/hip_bf16.h>
#include <cstdint>

// Problem constants
#define BB 4
#define TT 1024
#define CC 1024
#define HH 16
#define DD 64
#define MM (BB*TT)   // 4096

typedef __attribute__((ext_vector_type(16))) __bf16 v16bf;
typedef __attribute__((ext_vector_type(8)))  float  v8f;
typedef __attribute__((ext_vector_type(8)))  unsigned int v8u;
typedef __attribute__((ext_vector_type(4)))  unsigned int v4u;
typedef __attribute__((ext_vector_type(8)))  int v8i;
typedef __attribute__((ext_vector_type(4)))  int v4i;

union AFrag { v8u u; v16bf b; };
union Acc   { v8f v; float f[8]; };

__device__ __forceinline__ unsigned short f2bf(float x) {
    unsigned int u = __builtin_bit_cast(unsigned int, x);
    unsigned int r = (u + 0x7FFFu + ((u >> 16) & 1u)) >> 16;
    return (unsigned short)r;
}

__device__ __forceinline__ v8f wmma_bf16(v16bf a, v16bf b, v8f c) {
    return __builtin_amdgcn_wmma_f32_16x16x32_bf16(false, a, false, b, (short)0, c, false, false);
}

// ---- Tensor Data Mover: 2D bf16 tile load, global -> LDS, with LDS pad ----
// tile_w elems per row (row contiguous in memory), tile_h rows,
// src row stride = stride_elems, LDS pad: 1 DWORD after every 2^(pi+1) DWORDs.
__device__ __forceinline__ void tdm_load_2d(unsigned lds_off, const void* gaddr,
                                            unsigned tile_w, unsigned tile_h,
                                            unsigned stride_elems, unsigned pi)
{
    unsigned long long ga = (unsigned long long)(uintptr_t)gaddr;
    v4u g0;
    g0.x = 1u;                                                   // count=1, user mode
    g0.y = lds_off;                                              // LDS byte address
    g0.z = (unsigned)ga;                                         // global_addr[31:0]
    g0.w = ((unsigned)(ga >> 32) & 0x1FFFFFFu) | (2u << 30);     // addr[56:32] | type=2
    v8i g1;
    g1[0] = (int)((1u << 16)                                     // data_size = 2 bytes
                | (1u << 20)                                     // pad_enable
                | (pi << 22));                                   // pad_interval
    g1[1] = (int)((tile_w & 0xFFFFu) << 16);                     // tensor_dim0[15:0]
    g1[2] = (int)((tile_h & 0xFFFFu) << 16);                     // tensor_dim1[15:0]
    g1[3] = (int)((tile_w & 0xFFFFu) << 16);                     // tile_dim0
    g1[4] = (int)(tile_h & 0xFFFFu);                             // tile_dim1 (tile_dim2=0)
    g1[5] = (int)stride_elems;                                   // tensor_dim0_stride lo
    g1[6] = 0;
    g1[7] = 0;
    v4i z4 = {0, 0, 0, 0};
    v8i z8 = {0, 0, 0, 0, 0, 0, 0, 0};
    __builtin_amdgcn_tensor_load_to_lds(g0, g1, z4, z4, z8, 0);
}

__device__ __forceinline__ unsigned lds_off_of(const void* p) {
    return (unsigned)(uintptr_t)p;   // generic LDS pointer: low 32 bits = LDS offset
}

// ---------------- elementwise converters ----------------
__global__ void cvt_k(const float* __restrict__ in, unsigned short* __restrict__ out, int n) {
    int i = blockIdx.x * 256 + threadIdx.x;
    if (i < n) out[i] = f2bf(in[i]);
}

// Wq/Wk/Wv: [H,C,D] fp32 -> [C, H*D] bf16 row-major
__global__ void cvt_w_hcd_k(const float* __restrict__ in, unsigned short* __restrict__ out) {
    int i = blockIdx.x * 256 + threadIdx.x;       // 0 .. H*C*D-1 (1M)
    int h = i >> 16;                               // C*D = 65536
    int c = (i >> 6) & 1023;
    int d = i & 63;
    out[c * (HH * DD) + h * DD + d] = f2bf(in[i]);
}

// ---------------- tiled WMMA GEMM ----------------
// C[4096,1024] = A[4096,1024]bf16 x B[1024,1024]bf16.
// Block tile 128x128, Kstep 32, 8 waves as 4(M) x 2(N); wave tile 32x64 (8 WMMA/step).
// A tile staged by TDM (padded rows of 34 elems); B tile staged manually transposed.
template<int OUT_MODE>
__global__ __launch_bounds__(256)
void gemm_bf16_k(const unsigned short* __restrict__ A,
                 const unsigned short* __restrict__ Bm,
                 const float* __restrict__ bias,
                 unsigned short* __restrict__ outB,
                 float* __restrict__ outF)
{
    __shared__ unsigned short As[128 * 34];  // [m][k], row stride 34 (TDM pad 1 dword/16)
    __shared__ unsigned short Bs[128 * 36];  // transposed: [n][k], row stride 36

    const int tid  = threadIdx.x;
    const int lane = tid & 31, wid = tid >> 5;
    const int grp  = lane >> 4, l16 = lane & 15;
    const int wm = wid >> 1, wn = wid & 1;
    const int m0 = blockIdx.y * 128;
    const int n0 = blockIdx.x * 128;
    const unsigned ldsA = lds_off_of(As);

    Acc acc[2][4];
    #pragma unroll
    for (int mi = 0; mi < 2; mi++)
        #pragma unroll
        for (int i = 0; i < 4; i++)
            #pragma unroll
            for (int j = 0; j < 8; j++) acc[mi][i].f[j] = 0.0f;

    for (int k0 = 0; k0 < 1024; k0 += 32) {
        __syncthreads();
        if (wid == 0)   // TDM: A tile 128 rows x 32 k, src stride 1024
            tdm_load_2d(ldsA, A + (size_t)m0 * 1024 + k0, 32, 128, 1024, 3);
        { // stage B tile 32x128 transposed -> Bs[n][k]
            int kr = tid >> 3;               // k row 0..31
            int nb = (tid & 7) * 16;         // n col base
            const unsigned short* src = Bm + (size_t)(k0 + kr) * 1024 + n0 + nb;
            #pragma unroll
            for (int i = 0; i < 16; i++) Bs[(nb + i) * 36 + kr] = src[i];
        }
        if (wid == 0) __builtin_amdgcn_s_wait_tensorcnt(0);
        __syncthreads();

        AFrag a[2];
        #pragma unroll
        for (int mi = 0; mi < 2; mi++)
            #pragma unroll
            for (int v = 0; v < 8; v++) {
                int kk = ((v < 4) ? 2 * v : 2 * v + 8) + (grp ? 8 : 0);
                a[mi].u[v] = *(const unsigned int*)&As[(wm * 32 + mi * 16 + l16) * 34 + kk];
            }
        #pragma unroll
        for (int nt = 0; nt < 4; nt++) {
            AFrag bfrag;
            #pragma unroll
            for (int v = 0; v < 8; v++) {
                int kk = ((v < 4) ? 2 * v : 2 * v + 8) + 8 * grp;
                bfrag.u[v] = *(const unsigned int*)&Bs[(wn * 64 + nt * 16 + l16) * 36 + kk];
            }
            acc[0][nt].v = wmma_bf16(a[0].b, bfrag.b, acc[0][nt].v);
            acc[1][nt].v = wmma_bf16(a[1].b, bfrag.b, acc[1][nt].v);
        }
    }

    #pragma unroll
    for (int mi = 0; mi < 2; mi++)
        #pragma unroll
        for (int nt = 0; nt < 4; nt++)
            #pragma unroll
            for (int v = 0; v < 8; v++) {
                int m = m0 + wm * 32 + mi * 16 + v + 8 * grp;
                int n = n0 + wn * 64 + nt * 16 + l16;
                float val = acc[mi][nt].f[v];
                if (OUT_MODE == 0) {
                    int b = m >> 10, t = m & 1023;
                    int h = n >> 6,  d = n & 63;
                    outB[((size_t)(b * HH + h) * TT + t) * DD + d] = f2bf(val);
                } else {
                    outF[(size_t)m * 1024 + n] = val + bias[n];
                }
            }
}

// ---------------- flash attention ----------------
// grid (T/128, H, B); 8 waves; wave owns 16 query rows; key tiles of 64.
// K/V tiles staged by TDM (padded rows: 64 elems + 1 dword pad -> stride 66).
__global__ __launch_bounds__(256)
void attn_k(const unsigned short* __restrict__ Q,
            const unsigned short* __restrict__ Kp,
            const unsigned short* __restrict__ Vp,
            unsigned short* __restrict__ O)
{
    __shared__ unsigned short Ks[64 * 66];     // [key][d] stride 66
    __shared__ unsigned short Vs[64 * 66];
    __shared__ unsigned short Ps[8][16 * 64];  // per-wave P staging [row][key]

    const int tid  = threadIdx.x;
    const int lane = tid & 31, wid = tid >> 5;
    const int grp  = lane >> 4, l16 = lane & 15;
    const int qb = blockIdx.x * 128;
    const int h  = blockIdx.y, b = blockIdx.z;
    const size_t head_off = (size_t)(b * HH + h) * TT * DD;
    const unsigned ldsK = lds_off_of(Ks), ldsV = lds_off_of(Vs);

    // Q fragments for this wave's 16 rows (2 K-steps of 32 over D=64)
    unsigned int qf[2][8];
    {
        const unsigned short* qrow = Q + head_off + (size_t)(qb + wid * 16 + l16) * DD;
        #pragma unroll
        for (int kk = 0; kk < 2; kk++)
            #pragma unroll
            for (int v = 0; v < 8; v++) {
                int d = ((v < 4) ? 2 * v : 2 * v + 8) + (grp ? 8 : 0) + kk * 32;
                qf[kk][v] = *(const unsigned int*)&qrow[d];
            }
    }

    Acc Oacc[4];
    #pragma unroll
    for (int i = 0; i < 4; i++)
        #pragma unroll
        for (int j = 0; j < 8; j++) Oacc[i].f[j] = 0.0f;
    float mrow[8], lrow[8];
    #pragma unroll
    for (int v = 0; v < 8; v++) { mrow[v] = -1e30f; lrow[v] = 0.0f; }

    const int ntiles = qb / 64 + 2;   // causal: keys up to qb+127
    for (int j = 0; j < ntiles; ++j) {
        __syncthreads();
        if (wid == 0) {   // TDM: K and V 64x64 tiles
            tdm_load_2d(ldsK, Kp + head_off + (size_t)j * 64 * DD, 64, 64, 64, 4);
            tdm_load_2d(ldsV, Vp + head_off + (size_t)j * 64 * DD, 64, 64, 64, 4);
            __builtin_amdgcn_s_wait_tensorcnt(0);
        }
        __syncthreads();

        // S = Q * K^T
        Acc Sacc[4];
        #pragma unroll
        for (int i = 0; i < 4; i++)
            #pragma unroll
            for (int jj = 0; jj < 8; jj++) Sacc[i].f[jj] = 0.0f;
        #pragma unroll
        for (int kk = 0; kk < 2; kk++) {
            AFrag a;
            #pragma unroll
            for (int v = 0; v < 8; v++) a.u[v] = qf[kk][v];
            #pragma unroll
            for (int nt = 0; nt < 4; nt++) {
                AFrag bfrag;
                #pragma unroll
                for (int v = 0; v < 8; v++) {
                    int d = ((v < 4) ? 2 * v : 2 * v + 8) + 8 * grp + kk * 32;
                    bfrag.u[v] = *(const unsigned int*)&Ks[(nt * 16 + l16) * 66 + d];
                }
                Sacc[nt].v = wmma_bf16(a.b, bfrag.b, Sacc[nt].v);
            }
        }

        // scale, causal mask, online softmax; stage P (bf16) for PV
        #pragma unroll
        for (int v = 0; v < 8; v++) {
            int rg = qb + wid * 16 + v + 8 * grp;
            float s[4];
            float mloc = -1e30f;
            #pragma unroll
            for (int nt = 0; nt < 4; nt++) {
                int kg = j * 64 + nt * 16 + l16;
                float sv = Sacc[nt].f[v] * 0.125f;     // 1/sqrt(64)
                if (kg > rg) sv = -1e30f;
                s[nt] = sv;
                mloc = fmaxf(mloc, sv);
            }
            #pragma unroll
            for (int off = 8; off > 0; off >>= 1)
                mloc = fmaxf(mloc, __shfl_xor(mloc, off, 16));
            float mnew  = fmaxf(mrow[v], mloc);
            float alpha = __expf(mrow[v] - mnew);
            float rsum = 0.0f;
            #pragma unroll
            for (int nt = 0; nt < 4; nt++) {
                float p = __expf(s[nt] - mnew);
                rsum += p;
                Ps[wid][(v + 8 * grp) * 64 + nt * 16 + l16] = f2bf(p);
            }
            #pragma unroll
            for (int off = 8; off > 0; off >>= 1)
                rsum += __shfl_xor(rsum, off, 16);
            lrow[v] = lrow[v] * alpha + rsum;
            mrow[v] = mnew;
            #pragma unroll
            for (int nt = 0; nt < 4; nt++) Oacc[nt].f[v] *= alpha;
        }
        __syncthreads();

        // O += P * V
        #pragma unroll
        for (int kk = 0; kk < 2; kk++) {
            AFrag a;
            #pragma unroll
            for (int v = 0; v < 8; v++) {
                int kidx = ((v < 4) ? 2 * v : 2 * v + 8) + (grp ? 8 : 0) + kk * 32;
                a.u[v] = *(const unsigned int*)&Ps[wid][l16 * 64 + kidx];
            }
            #pragma unroll
            for (int nt = 0; nt < 4; nt++) {
                AFrag bfrag;
                #pragma unroll
                for (int v = 0; v < 8; v++) {
                    int krow = ((v < 4) ? 2 * v : 2 * v + 8) + 8 * grp + kk * 32;
                    unsigned int lo = Vs[krow * 66 + nt * 16 + l16];
                    unsigned int hi = Vs[(krow + 1) * 66 + nt * 16 + l16];
                    bfrag.u[v] = lo | (hi << 16);
                }
                Oacc[nt].v = wmma_bf16(a.b, bfrag.b, Oacc[nt].v);
            }
        }
    }

    // write att out as [B, T, H, D]  (== row-major [4096, 1024] for final GEMM)
    #pragma unroll
    for (int nt = 0; nt < 4; nt++)
        #pragma unroll
        for (int v = 0; v < 8; v++) {
            int t = qb + wid * 16 + v + 8 * grp;
            int d = nt * 16 + l16;
            float val = Oacc[nt].f[v] / lrow[v];
            O[((size_t)(b * TT + t) * HH + h) * DD + d] = f2bf(val);
        }
}

// ---------------- host ----------------
extern "C" void kernel_launch(void* const* d_in, const int* in_sizes, int n_in,
                              void* d_out, int out_size, void* d_ws, size_t ws_size,
                              hipStream_t stream) {
    const float* x  = (const float*)d_in[0];
    const float* Wq = (const float*)d_in[1];
    const float* Wk = (const float*)d_in[2];
    const float* Wv = (const float*)d_in[3];
    const float* Wo = (const float*)d_in[4];
    const float* bo = (const float*)d_in[5];
    float* out = (float*)d_out;

    char* ws = (char*)d_ws;
    const size_t MB = 1u << 20;
    unsigned short* xb  = (unsigned short*)(ws);             // 8 MB: x bf16 [4096,1024]
    unsigned short* wqb = (unsigned short*)(ws +  8 * MB);   // 2 MB: Wq bf16 [C, H*D]
    unsigned short* wkb = (unsigned short*)(ws + 10 * MB);
    unsigned short* wvb = (unsigned short*)(ws + 12 * MB);
    unsigned short* wob = (unsigned short*)(ws + 14 * MB);   // 2 MB: Wo bf16 [1024,1024]
    unsigned short* qws = (unsigned short*)(ws + 16 * MB);   // 8 MB: q [B,H,T,D]
    unsigned short* kws = (unsigned short*)(ws + 24 * MB);
    unsigned short* vws = (unsigned short*)(ws + 32 * MB);
    unsigned short* att = (unsigned short*)(ws + 40 * MB);   // 8 MB: att [B,T,H*D]

    // converts
    cvt_k<<<(MM * CC) / 256, 256, 0, stream>>>(x, xb, MM * CC);
    cvt_w_hcd_k<<<(HH * CC * DD) / 256, 256, 0, stream>>>(Wq, wqb);
    cvt_w_hcd_k<<<(HH * CC * DD) / 256, 256, 0, stream>>>(Wk, wkb);
    cvt_w_hcd_k<<<(HH * CC * DD) / 256, 256, 0, stream>>>(Wv, wvb);
    cvt_k<<<(CC * CC) / 256, 256, 0, stream>>>(Wo, wob, CC * CC);

    // QKV projections: [4096,1024] x [1024,1024] -> [B,H,T,D] bf16
    dim3 gg(1024 / 128, MM / 128);
    gemm_bf16_k<0><<<gg, 256, 0, stream>>>(xb, wqb, nullptr, qws, nullptr);
    gemm_bf16_k<0><<<gg, 256, 0, stream>>>(xb, wkb, nullptr, kws, nullptr);
    gemm_bf16_k<0><<<gg, 256, 0, stream>>>(xb, wvb, nullptr, vws, nullptr);

    // flash attention
    dim3 ga(TT / 128, HH, BB);
    attn_k<<<ga, 256, 0, stream>>>(qws, kws, vws, att);

    // output projection + bias -> fp32 d_out
    gemm_bf16_k<1><<<gg, 256, 0, stream>>>(att, wob, bo, nullptr, out);
}
